// LearnableDemosaick_34797825032630
// MI455X (gfx1250) — compile-verified, matrix operand order checked by reading, not run on hardware
//
#include <hip/hip_runtime.h>
#include <cstdint>

// Problem constants (reference: B,H,W,K,F = 16,512,512,8,5)
#define BB 16
#define HH 512
#define WW 512
#define KK 8
#define FF 5

typedef __attribute__((ext_vector_type(16))) _Float16 v16h;
typedef __attribute__((ext_vector_type(8)))  float    v8f;
typedef __attribute__((ext_vector_type(8)))  uint32_t v8u;

// Tile geometry: 32x32 pixels per block, 2-pixel halo each side.
#define TY 32
#define TX 32
#define TROWS (TY + 4)           // 36
#define TCOLS (TX + 4)           // 36
#define TSTRIDE 40               // padded u16 row stride (bank spread)
#define ZOFF   (TROWS * TSTRIDE) // 1440: start of always-zero pad region
#define ZPAD   1280              // covers base offsets [0,1271] past ZOFF
#define SMSTRIDE 20              // padded f32 row stride for softmax buffer

__global__ __launch_bounds__(256)
void demosaick_green_wmma(const float* __restrict__ mosaick,
                          const float* __restrict__ sel_filts,
                          const float* __restrict__ green_filts,
                          float* __restrict__ out) {
    // f16 halo tile stored as raw u16 bit patterns (+ zero pad region for padded taps).
    __shared__ uint16_t tile[ZOFF + ZPAD];                     // 5440 B
    __shared__ __align__(16) float smax[8 * TY * SMSTRIDE];    // 8 waves * 32 rows * 20 f32

    const int tid  = threadIdx.x;
    const int lane = tid & 31;
    const int wid  = tid >> 5;
    const int hi   = lane >> 4;      // 0: lanes 0-15, 1: lanes 16-31
    const int m    = lane & 15;      // pixel-in-segment (A rows) / filter column (B)

    const int b  = blockIdx.z;
    const int y0 = blockIdx.y * TY;
    const int x0 = blockIdx.x * TX;

    const float* img = mosaick + (size_t)b * (HH * WW);

    // ---------------- Build B (filters) in registers, once per lane ----------------
    // Column n: 0-7 = masked sel filter n, 8-15 = masked green filter n-8.
    // 16-bit B layout mirrors A: lanes 0-15 hold K 0-7 / 16-23, lanes 16-31 hold K 8-15 / 24-31.
    const float* filt = (m < 8) ? (sel_filts + m * (FF * FF))
                                : (green_filts + (m - 8) * (FF * FF));
    v16h bmat;
#pragma unroll
    for (int p = 0; p < 16; ++p) {
        const int  kLo = (p < 8) ? p : (p + 8);          // k for lanes 0-15 (always < 25)
        const int  kHi = kLo + 8;                        // k for lanes 16-31
        const bool vLo = (((kLo / FF) + (kLo % FF)) & 1) == 1;                    // checkerboard mask
        const bool vHi = (kHi < FF * FF) && ((((kHi / FF) + (kHi % FF)) & 1) == 1);
        const int  k   = hi ? kHi : kLo;
        const bool ok  = hi ? vHi : vLo;
        float v = 0.0f;
        if (ok) v = filt[k];                              // masked / zero-padded tap
        bmat[p] = (_Float16)v;
    }

    // ---- Loop-invariant A-gather offsets (u16-element units). Computed ONCE. ----
    // Padded taps (k>=25, hi lanes only) point into the zeroed LDS region, so the hot
    // loop is pure ds_load_u16 with immediate offsets: no masking VALU.
    int aoff[16];
#pragma unroll
    for (int p = 0; p < 16; ++p) {
        const int kLo = (p < 8) ? p : (p + 8);
        const int kHi = kLo + 8;
        const int oLo = (kLo / FF) * TSTRIDE + (kLo % FF);
        const int oHi = (kHi < FF * FF) ? ((kHi / FF) * TSTRIDE + (kHi % FF)) : ZOFF;
        aoff[p] = wid * TSTRIDE + m + (hi ? oHi : oLo);
    }

    // ---------------- Stage halo tile into LDS as f16 bits (zero pad at borders) ---
    for (int idx = tid; idx < TROWS * TCOLS; idx += 256) {
        const int r  = idx / TCOLS;
        const int c  = idx - r * TCOLS;
        const int gy = y0 - 2 + r;
        const int gx = x0 - 2 + c;
        float v = 0.0f;
        if (gy >= 0 && gy < HH && gx >= 0 && gx < WW) v = img[gy * WW + gx];
        tile[r * TSTRIDE + c] = __builtin_bit_cast(uint16_t, (_Float16)v);
    }
    // Zero region used to synthesize the K=25..31 zero taps of A.
    for (int idx = tid; idx < ZPAD; idx += 256) {
        tile[ZOFF + idx] = 0;
    }
    __syncthreads();

    float* sm = &smax[wid * (TY * SMSTRIDE)];

    // Row mapping yl = i*8 + wid: the i/s-dependent part of every LDS A-address is a
    // compile-time constant (folds into the ds_load immediate); the VGPR part is aoff[p].
#pragma unroll
    for (int i = 0; i < 4; ++i) {
        const int yl = i * 8 + wid;                      // local row 0..31

#pragma unroll
        for (int s = 0; s < 2; ++s) {
            const int cimm = i * 8 * TSTRIDE + s * 16;   // compile-time immediate part

            // ------- Build A: 16 pixels x 32 taps -------
            // Phase 1: issue all 16 independent ds_load_u16 so the whole batch is in
            // flight before any consumer waits (staggered s_wait_dscnt pipelining).
            uint32_t t[16];
#pragma unroll
            for (int p = 0; p < 16; ++p) {
                t[p] = tile[cimm + aoff[p]];
            }
            // Phase 2: pack two taps per VGPR (A layout: half pairs per 32-bit reg).
            v8u apk;
#pragma unroll
            for (int j = 0; j < 8; ++j) {
                apk[j] = t[2 * j] | (t[2 * j + 1] << 16);
            }
            const v16h a = __builtin_bit_cast(v16h, apk);

            // ------- One WMMA: D[pixel 0..15][sel0..7 | grn0..7] -------
            v8f c = {};
            v8f d = __builtin_amdgcn_wmma_f32_16x16x32_f16(
                false, a, false, bmat, (short)0, c, false, false);

            // ------- Scatter D to per-wave softmax buffer -------
            // VGPR r, lane: D[M][N] with M = r + 8*hi, N = lane&15; buffer row = local x.
#pragma unroll
            for (int r = 0; r < 8; ++r) {
                sm[(s * 16 + r + hi * 8) * SMSTRIDE + m] = d[r];
            }
        }

        // ------- Per-pixel softmax combine + checkerboard output -------
        // Lane l owns local pixel (yl, l): 16 scores contiguous in its buffer row.
        const float4 q0 = *reinterpret_cast<const float4*>(&sm[lane * SMSTRIDE + 0]);   // sel 0-3
        const float4 q1 = *reinterpret_cast<const float4*>(&sm[lane * SMSTRIDE + 4]);   // sel 4-7
        const float4 q2 = *reinterpret_cast<const float4*>(&sm[lane * SMSTRIDE + 8]);   // grn 0-3
        const float4 q3 = *reinterpret_cast<const float4*>(&sm[lane * SMSTRIDE + 12]);  // grn 4-7

        float mx = fmaxf(fmaxf(fmaxf(q0.x, q0.y), fmaxf(q0.z, q0.w)),
                         fmaxf(fmaxf(q1.x, q1.y), fmaxf(q1.z, q1.w)));
        float e0 = __expf(q0.x - mx), e1 = __expf(q0.y - mx);
        float e2 = __expf(q0.z - mx), e3 = __expf(q0.w - mx);
        float e4 = __expf(q1.x - mx), e5 = __expf(q1.y - mx);
        float e6 = __expf(q1.z - mx), e7 = __expf(q1.w - mx);
        float den = (e0 + e1) + (e2 + e3) + ((e4 + e5) + (e6 + e7));
        float num = e0 * q2.x + e1 * q2.y + e2 * q2.z + e3 * q2.w
                  + e4 * q3.x + e5 * q3.y + e6 * q3.z + e7 * q3.w;
        const float interp = num / den;

        const int gy = y0 + yl;
        const int gx = x0 + lane;
        const float raw = img[gy * WW + gx];                       // exact f32 green sample
        const float gvn = (((gy + gx) & 1) == 0) ? raw : interp;   // GRBG: green where (y+x) even
        out[((size_t)b * HH + gy) * WW + gx] = gvn;
    }
}

extern "C" void kernel_launch(void* const* d_in, const int* in_sizes, int n_in,
                              void* d_out, int out_size, void* d_ws, size_t ws_size,
                              hipStream_t stream) {
    (void)in_sizes; (void)n_in; (void)d_ws; (void)ws_size; (void)out_size;
    const float* mosaick     = (const float*)d_in[0];   // [16,1,512,512] f32
    const float* sel_filts   = (const float*)d_in[1];   // [8,5,5] f32
    const float* green_filts = (const float*)d_in[2];   // [8,5,5] f32
    float* out = (float*)d_out;                         // [16,1,512,512] f32

    dim3 grid(WW / TX, HH / TY, BB);   // (16,16,16)
    dim3 block(256);
    demosaick_green_wmma<<<grid, block, 0, stream>>>(mosaick, sel_filts, green_filts, out);
}
